// GraphConvolution_75788992905409
// MI455X (gfx1250) — matrix operand order repaired and checked by reading, not compiled, not run on hardware
//
#include <hip/hip_runtime.h>
#include <hip/hip_bf16.h>

typedef __attribute__((ext_vector_type(16))) _Float16 v16h;
typedef __attribute__((ext_vector_type(8)))  _Float16 v8h;
typedef __attribute__((ext_vector_type(8)))  float    v8f;

#define WMMA(a, b, c) \
  __builtin_amdgcn_wmma_f32_16x16x32_f16(false, (a), false, (b), (short)0, (c), false, false)

#define LOG2E_F 1.44269504088896f
#define LN2_F   0.69314718055995f

// Fast activations on raw CDNA5 transcendentals (v_exp_f32 / v_log_f32 / v_rcp_f32).
__device__ __forceinline__ float fast_sigmoid(float x) {
  float e = __builtin_amdgcn_exp2f(-x * LOG2E_F);
  return __builtin_amdgcn_rcpf(1.0f + e);
}
__device__ __forceinline__ float fast_softplus(float x) {
  float e = __builtin_amdgcn_exp2f(-fabsf(x) * LOG2E_F);
  return fmaxf(x, 0.0f) + __builtin_amdgcn_logf(1.0f + e) * LN2_F;
}

// ---------------------------------------------------------------------------
// Fragment layout (wave32; A/B use contiguous k, this code's choice, not the ISA table order):
//  A 16x32 f16: lane row = lane&15; elem j -> K = kbase + 16*half + j   (contiguous)
//  B 32x16 f16: lane col = lane&15; elem j -> K = kbase + 16*half + j   (contiguous)
//  C/D:         lane l, vgpr r -> (M = r + 8*half, N = l&15)
// ---------------------------------------------------------------------------

__device__ __forceinline__ v16h concat8(v8h lo, v8h hi) {
  return __builtin_shufflevector(lo, hi, 0, 1, 2, 3, 4, 5, 6, 7,
                                 8, 9, 10, 11, 12, 13, 14, 15);
}

__device__ __forceinline__ v16h load_A16(const _Float16* __restrict__ row, int kbase, int half) {
  return *(const v16h*)(row + kbase + half * 16);       // 32-byte aligned
}

__device__ __forceinline__ v16h load_B16(const _Float16* __restrict__ Wt_row, int kbase, int half) {
  return *(const v16h*)(Wt_row + kbase + half * 16);    // 32-byte aligned
}

// ---------------------------------------------------------------------------
// One-time converters
// ---------------------------------------------------------------------------
__global__ void f32_to_f16_pad_rows_kernel(const float* __restrict__ in, _Float16* __restrict__ out,
                                           int rows, int Kin, int Kout) {
  for (long long idx = (long long)blockIdx.x * blockDim.x + threadIdx.x;
       idx < (long long)rows * Kout; idx += (long long)gridDim.x * blockDim.x) {
    int r = (int)(idx / Kout), k = (int)(idx % Kout);
    out[idx] = (k < Kin) ? (_Float16)in[(size_t)r * Kin + k] : (_Float16)0.0f;
  }
}

// W [K x 64] f32 row-major -> Wt [64 x Kpad] f16 (zero padded)
__global__ void w_transpose_f16_kernel(const float* __restrict__ W, _Float16* __restrict__ Wt,
                                       int K, int Kpad) {
  for (int idx = blockIdx.x * blockDim.x + threadIdx.x; idx < 64 * Kpad;
       idx += gridDim.x * blockDim.x) {
    int n = idx / Kpad, k = idx % Kpad;
    Wt[idx] = (k < K) ? (_Float16)W[(size_t)k * 64 + n] : (_Float16)0.0f;
  }
}

// ---------------------------------------------------------------------------
// out[nrows,64] = relu(in16[nrows,KPAD] @ W + bias)
// ---------------------------------------------------------------------------
template <int KPAD, bool WRITE32>
__global__ __launch_bounds__(128) void linear_relu_f16_kernel(
    const _Float16* __restrict__ in16, const _Float16* __restrict__ Wt,
    const float* __restrict__ bias, _Float16* __restrict__ out16,
    float* __restrict__ out32, int nrows) {
  const int lane = threadIdx.x & 31;
  const int wv   = threadIdx.x >> 5;
  const int q    = lane & 15;
  const int half = lane >> 4;
  const int n    = wv * 16 + q;
  constexpr int NK = KPAD / 32;

  v16h B[NK];
  const _Float16* wrow = Wt + (size_t)n * KPAD;
#pragma unroll
  for (int c = 0; c < NK; ++c) B[c] = load_B16(wrow, c * 32, half);
  const float bn = bias[n];

  const int ntiles = (nrows + 15) >> 4;
  for (int t = blockIdx.x; t < ntiles; t += gridDim.x) {
    int row = t * 16 + q;
    if (row >= nrows) row = nrows - 1;                  // clamp; EXEC stays full
    const _Float16* rp = in16 + (size_t)row * KPAD;

    v8f acc = {};
#pragma unroll
    for (int c = 0; c < NK; ++c) acc = WMMA(load_A16(rp, c * 32, half), B[c], acc);

#pragma unroll
    for (int r = 0; r < 8; ++r) {
      int m = t * 16 + r + half * 8;
      if (m < nrows) {
        float v = fmaxf(acc[r] + bn, 0.0f);
        out16[(size_t)m * 64 + n] = (_Float16)v;
        if (WRITE32) out32[(size_t)m * 64 + n] = v;
      }
    }
  }
}

// ---------------------------------------------------------------------------
// CGConv edge messages: m = sigmoid(z@Wf+bf) * softplus(z@Ws+bs),
// z = [h[dst] || h[src] || ea]; atomic-scatter into agg[dst].
// ---------------------------------------------------------------------------
__global__ __launch_bounds__(128) void cgconv_edge_wmma_kernel(
    const _Float16* __restrict__ h16, const int* __restrict__ src,
    const int* __restrict__ dst, const float* __restrict__ ea,
    const _Float16* __restrict__ WfT, const _Float16* __restrict__ WsT,
    const float* __restrict__ Wf, const float* __restrict__ Ws,   // f32 [129,64]
    const float* __restrict__ bf, const float* __restrict__ bs,
    float* __restrict__ agg, int E) {
  const int lane = threadIdx.x & 31;
  const int wv   = threadIdx.x >> 5;
  const int q    = lane & 15;
  const int half = lane >> 4;
  const int n    = wv * 16 + q;

  const _Float16* wfr = WfT + (size_t)n * 128;
  const _Float16* wsr = WsT + (size_t)n * 128;
  v16h BF0 = load_B16(wfr, 0, half),  BF1 = load_B16(wfr, 32, half);
  v16h BF2 = load_B16(wfr, 64, half), BF3 = load_B16(wfr, 96, half);
  v16h BS0 = load_B16(wsr, 0, half),  BS1 = load_B16(wsr, 32, half);
  v16h BS2 = load_B16(wsr, 64, half), BS3 = load_B16(wsr, 96, half);
  const float biasF = bf[n], biasS = bs[n];
  const float wlF = Wf[(size_t)128 * 64 + n], wlS = Ws[(size_t)128 * 64 + n];

  const int etiles = (E + 15) >> 4;
  for (int t = blockIdx.x; t < etiles; t += gridDim.x) {
    const int e0 = t * 16;
    int eA = e0 + q;
    if (eA >= E) eA = E - 1;
    const _Float16* hd = h16 + (size_t)dst[eA] * 64;
    const _Float16* hs = h16 + (size_t)src[eA] * 64;

    v16h Ad0 = load_A16(hd, 0, half), Ad1 = load_A16(hd, 32, half);
    v16h As0 = load_A16(hs, 0, half), As1 = load_A16(hs, 32, half);

    v8f accF = {}, accS = {};
    accF = WMMA(Ad0, BF0, accF);
    accF = WMMA(Ad1, BF1, accF);
    accF = WMMA(As0, BF2, accF);
    accF = WMMA(As1, BF3, accF);
    accS = WMMA(Ad0, BS0, accS);
    accS = WMMA(Ad1, BS1, accS);
    accS = WMMA(As0, BS2, accS);
    accS = WMMA(As1, BS3, accS);

#pragma unroll
    for (int r = 0; r < 8; ++r) {
      int e = e0 + r + half * 8;
      if (e < E) {
        float eav = ea[e];
        float f = accF[r] + biasF + eav * wlF;
        float s = accS[r] + biasS + eav * wlS;
        unsafeAtomicAdd(&agg[(size_t)dst[e] * 64 + n], fast_sigmoid(f) * fast_softplus(s));
      }
    }
  }
}

// ---------------------------------------------------------------------------
// Elementwise / scatter helpers
// ---------------------------------------------------------------------------
__global__ void zero_kernel(float* __restrict__ p, size_t nf) {
  for (size_t i = (size_t)blockIdx.x * blockDim.x + threadIdx.x; i < nf;
       i += (size_t)gridDim.x * blockDim.x)
    p[i] = 0.0f;
}

__global__ void degree_kernel(const int* __restrict__ dst, float* __restrict__ deg, int E) {
  for (int e = blockIdx.x * blockDim.x + threadIdx.x; e < E; e += gridDim.x * blockDim.x)
    unsafeAtomicAdd(&deg[dst[e]], 1.0f);
}

__global__ void clamp1_kernel(float* __restrict__ deg, int N) {
  for (int i = blockIdx.x * blockDim.x + threadIdx.x; i < N; i += gridDim.x * blockDim.x)
    deg[i] = fmaxf(deg[i], 1.0f);
}

__global__ void node_update_kernel(float* __restrict__ h32, _Float16* __restrict__ h16,
                                   float* __restrict__ agg, const float* __restrict__ denom,
                                   int total) {
  for (int idx = blockIdx.x * blockDim.x + threadIdx.x; idx < total;
       idx += gridDim.x * blockDim.x) {
    float v = fmaxf(h32[idx] + agg[idx] * __builtin_amdgcn_rcpf(denom[idx >> 6]), 0.0f);
    h32[idx] = v;
    h16[idx] = (_Float16)v;
    agg[idx] = 0.0f;
  }
}

__global__ void pool_sum_kernel(const float* __restrict__ h, const int* __restrict__ bm,
                                float* __restrict__ gsum, int total) {
  for (int idx = blockIdx.x * blockDim.x + threadIdx.x; idx < total;
       idx += gridDim.x * blockDim.x)
    unsafeAtomicAdd(&gsum[(size_t)bm[idx >> 6] * 64 + (idx & 63)], h[idx]);
}

__global__ void pool_cnt_kernel(const int* __restrict__ bm, float* __restrict__ gcnt, int N) {
  for (int i = blockIdx.x * blockDim.x + threadIdx.x; i < N; i += gridDim.x * blockDim.x)
    unsafeAtomicAdd(&gcnt[bm[i]], 1.0f);
}

// ---------------------------------------------------------------------------
// Post-MLP: one block, 8 waves, B<=128 rows.
// ---------------------------------------------------------------------------
__global__ __launch_bounds__(256) void post_mlp_kernel(
    const float* __restrict__ gsum, const float* __restrict__ gcnt,
    const _Float16* __restrict__ Wt0, const float* __restrict__ b0,
    const _Float16* __restrict__ Wt1, const float* __restrict__ b1,
    const _Float16* __restrict__ Wt2, const float* __restrict__ b2,
    const float* __restrict__ Wfin, const float* __restrict__ bfin,
    float* __restrict__ out, int B) {
  __shared__ _Float16 bufA[128 * 64];
  __shared__ _Float16 bufB[128 * 64];
  const int tid = threadIdx.x;

  for (int idx = tid; idx < 128 * 64; idx += blockDim.x) {
    int rowi = idx >> 6;
    float v = 0.0f;
    if (rowi < B) v = gsum[idx] / fmaxf(gcnt[rowi], 1.0f);
    bufA[idx] = (_Float16)v;
  }
  __syncthreads();

  const int lane = tid & 31;
  const int wv   = tid >> 5;
  const int q    = lane & 15;
  const int half = lane >> 4;
  const int row0 = wv * 16 + q;

  const _Float16* Wl[3] = {Wt0, Wt1, Wt2};
  const float*    bl[3] = {b0, b1, b2};
  _Float16* sbuf = bufA;
  _Float16* dbuf = bufB;

  for (int l = 0; l < 3; ++l) {
    const _Float16* rp = sbuf + row0 * 64;
    v16h A0 = load_A16(rp, 0, half);
    v16h A1 = load_A16(rp, 32, half);
#pragma unroll
    for (int nt = 0; nt < 4; ++nt) {
      int n = nt * 16 + q;
      const _Float16* wrow = Wl[l] + (size_t)n * 64;
      v8f acc = {};
      acc = WMMA(A0, load_B16(wrow, 0, half), acc);
      acc = WMMA(A1, load_B16(wrow, 32, half), acc);
      float bn = bl[l][n];
#pragma unroll
      for (int r = 0; r < 8; ++r)
        dbuf[(wv * 16 + r + half * 8) * 64 + n] = (_Float16)fmaxf(acc[r] + bn, 0.0f);
    }
    __syncthreads();
    _Float16* tsw = sbuf; sbuf = dbuf; dbuf = tsw;
  }

  if (tid < B) {
    float sum = 0.0f;
#pragma unroll
    for (int k = 0; k < 64; ++k) sum += (float)sbuf[tid * 64 + k] * Wfin[k];
    out[tid] = sum + bfin[0];
  }
}

// ---------------------------------------------------------------------------
// Launch
// ---------------------------------------------------------------------------
static inline int imin(int a, int b) { return a < b ? a : b; }

extern "C" void kernel_launch(void* const* d_in, const int* in_sizes, int n_in,
                              void* d_out, int out_size, void* d_ws, size_t ws_size,
                              hipStream_t stream) {
  (void)n_in; (void)ws_size;
  const float* X         = (const float*)d_in[0];
  const int*   edge_idx  = (const int*)d_in[1];
  const float* edge_attr = (const float*)d_in[3];
  const int*   batch_map = (const int*)d_in[4];
  const float* pre_W0 = (const float*)d_in[5];
  const float* pre_b0 = (const float*)d_in[6];
  const float* pre_W1 = (const float*)d_in[7];
  const float* pre_b1 = (const float*)d_in[8];
  const float* pre_W2 = (const float*)d_in[9];
  const float* pre_b2 = (const float*)d_in[10];
  const float* cg_Wf  = (const float*)d_in[11];
  const float* cg_bf  = (const float*)d_in[12];
  const float* cg_Ws  = (const float*)d_in[13];
  const float* cg_bs  = (const float*)d_in[14];
  const float* post_W0 = (const float*)d_in[15];
  const float* post_b0 = (const float*)d_in[16];
  const float* post_W1 = (const float*)d_in[17];
  const float* post_b1 = (const float*)d_in[18];
  const float* post_W2 = (const float*)d_in[19];
  const float* post_b2 = (const float*)d_in[20];
  const float* post_Wf = (const float*)d_in[21];
  const float* post_bf = (const float*)d_in[22];
  float* out = (float*)d_out;

  const int N = in_sizes[4];
  const int E = in_sizes[3];
  const int F = in_sizes[0] / N;        // 92
  const int D = in_sizes[6];            // 64
  const int Z = 2 * D + 1;              // 129
  const int L = in_sizes[12] / D;       // 3
  const int B = out_size;               // 128
  const int FP = 96;                    // F padded to 3*32

  // ---- workspace layout (all region sizes are multiples of 32 bytes) ----
  char* p = (char*)d_ws;
  float* h32  = (float*)p;  p += (size_t)N * 64 * 4;
  float* agg  = (float*)p;  p += (size_t)N * 64 * 4;   // agg|deg|gsum|gcnt contiguous
  float* deg  = (float*)p;  p += (size_t)N * 4;
  float* gsum = (float*)p;  p += (size_t)B * 64 * 4;
  float* gcnt = (float*)p;  p += (size_t)((B + 7) & ~7) * 4;
  _Float16* h16   = (_Float16*)p; p += (size_t)N * 64 * 2;
  _Float16* t16a  = (_Float16*)p; p += (size_t)N * 64 * 2;
  _Float16* t16b  = (_Float16*)p; p += (size_t)N * 64 * 2;
  _Float16* Xh    = (_Float16*)p; p += (size_t)N * FP * 2;
  _Float16* WtP0  = (_Float16*)p; p += (size_t)64 * FP * 2;
  _Float16* WtP1  = (_Float16*)p; p += (size_t)64 * 64 * 2;
  _Float16* WtP2  = (_Float16*)p; p += (size_t)64 * 64 * 2;
  _Float16* WtO0  = (_Float16*)p; p += (size_t)64 * 64 * 2;
  _Float16* WtO1  = (_Float16*)p; p += (size_t)64 * 64 * 2;
  _Float16* WtO2  = (_Float16*)p; p += (size_t)64 * 64 * 2;
  _Float16* WtF0  = (_Float16*)p; p += (size_t)64 * 128 * 2;
  _Float16* WtS0  = (_Float16*)p;
  _Float16* WtF[8]; _Float16* WtS[8];
  for (int l = 0; l < L; ++l) { WtF[l] = WtF0 + (size_t)l * 64 * 128; }
  WtS0 = WtF0 + (size_t)L * 64 * 128;
  for (int l = 0; l < L; ++l) { WtS[l] = WtS0 + (size_t)l * 64 * 128; }

  const int* srcI = edge_idx;
  const int* dstI = edge_idx + E;

  const int ntiles = (N + 15) >> 4;
  const int etiles = (E + 15) >> 4;
  const int gridL  = imin(ntiles, 8192);
  const int gridE  = imin(etiles, 8192);
  const int gridN  = imin((N * 64 + 255) / 256, 4096);
  const int gridEl = imin((E + 255) / 256, 4096);
  const int gridNn = imin((N + 255) / 256, 4096);
  const int gridW  = 32;

  // ---- one-time f16 conversions / transposes ----
  f32_to_f16_pad_rows_kernel<<<4096, 256, 0, stream>>>(X, Xh, N, F, FP);
  w_transpose_f16_kernel<<<gridW, 256, 0, stream>>>(pre_W0, WtP0, F, FP);
  w_transpose_f16_kernel<<<gridW, 256, 0, stream>>>(pre_W1, WtP1, 64, 64);
  w_transpose_f16_kernel<<<gridW, 256, 0, stream>>>(pre_W2, WtP2, 64, 64);
  w_transpose_f16_kernel<<<gridW, 256, 0, stream>>>(post_W0, WtO0, 64, 64);
  w_transpose_f16_kernel<<<gridW, 256, 0, stream>>>(post_W1, WtO1, 64, 64);
  w_transpose_f16_kernel<<<gridW, 256, 0, stream>>>(post_W2, WtO2, 64, 64);
  for (int l = 0; l < L; ++l) {
    w_transpose_f16_kernel<<<gridW, 256, 0, stream>>>(cg_Wf + (size_t)l * Z * D, WtF[l], 128, 128);
    w_transpose_f16_kernel<<<gridW, 256, 0, stream>>>(cg_Ws + (size_t)l * Z * D, WtS[l], 128, 128);
  }

  // ---- pre-MLP ----
  linear_relu_f16_kernel<96, false><<<gridL, 128, 0, stream>>>(Xh,  WtP0, pre_b0, t16a, (float*)nullptr, N);
  linear_relu_f16_kernel<64, false><<<gridL, 128, 0, stream>>>(t16a, WtP1, pre_b1, t16b, (float*)nullptr, N);
  linear_relu_f16_kernel<64, true ><<<gridL, 128, 0, stream>>>(t16b, WtP2, pre_b2, h16, h32, N);

  // ---- zero agg|deg|gsum|gcnt (contiguous) ----
  size_t nz = (size_t)N * 64 + (size_t)N + (size_t)B * 64 + (size_t)((B + 7) & ~7);
  zero_kernel<<<4096, 256, 0, stream>>>(agg, nz);

  // ---- degree -> denom ----
  degree_kernel<<<gridEl, 256, 0, stream>>>(dstI, deg, E);
  clamp1_kernel<<<gridNn, 256, 0, stream>>>(deg, N);

  // ---- CGConv layers ----
  for (int l = 0; l < L; ++l) {
    cgconv_edge_wmma_kernel<<<gridE, 128, 0, stream>>>(
        h16, srcI, dstI, edge_attr, WtF[l], WtS[l],
        cg_Wf + (size_t)l * Z * D, cg_Ws + (size_t)l * Z * D,
        cg_bf + (size_t)l * D, cg_bs + (size_t)l * D, agg, E);
    node_update_kernel<<<gridN, 256, 0, stream>>>(h32, h16, agg, deg, N * 64);
  }

  // ---- global mean pool + post-MLP ----
  pool_sum_kernel<<<gridN, 256, 0, stream>>>(h32, batch_map, gsum, N * 64);
  pool_cnt_kernel<<<gridNn, 256, 0, stream>>>(batch_map, gcnt, N);
  post_mlp_kernel<<<1, 256, 0, stream>>>(gsum, gcnt, WtO0, post_b0, WtO1, post_b1,
                                         WtO2, post_b2, post_Wf, post_bf, out, B);
}
